// Decoder_83751862272699
// MI455X (gfx1250) — matrix operand order repaired and checked by reading, not compiled, not run on hardware
//
#include <hip/hip_runtime.h>
#include <hip/hip_bf16.h>
#include <math.h>

// ---------------------------------------------------------------------------
// Types for CDNA5 WMMA (gfx1250, wave32)
// ---------------------------------------------------------------------------
typedef __attribute__((ext_vector_type(16))) __bf16 v16bf;
typedef __attribute__((ext_vector_type(8)))  __bf16 v8bf;
typedef __attribute__((ext_vector_type(8)))  float  v8f;

#define HIDDEN 512
#define EMBED  50
#define VOCAB  83
#define TSEQ   600
#define BATCH  256

// Score-kernel LDS geometry (bytes)
#define A_CSTR    1088                 // per (mt,kc) A chunk: 1024 data + 64 pad
#define A_REGION  (64 * A_CSTR)        // 4 mt x 16 kc            = 69632
#define B_PSTR    512                  // piece stride inside a B chunk
#define B_CSTR    1088                 // per kc B chunk: 1024 data + 64 pad
#define B_BUF     (16 * B_CSTR)        // one nt worth of B       = 17408
#define SCORE_LDS (A_REGION + 2 * B_BUF)  // 104448

// f32 -> bf16 round-to-nearest-even (bit trick, NaN edge ignored)
static __device__ __forceinline__ unsigned int f32_to_bf16(float f) {
    unsigned int u = __float_as_uint(f);
    u += 0x7FFFu + ((u >> 16) & 1u);
    return u >> 16;
}

// ---------------------------------------------------------------------------
// CDNA5 async global->LDS copy (ASYNCcnt-tracked) + waits
// ---------------------------------------------------------------------------
static __device__ __forceinline__ void async_g2l_b128(unsigned lds_byte, const void* gaddr) {
    asm volatile("global_load_async_to_lds_b128 %0, %1, off"
                 :: "v"(lds_byte), "v"(gaddr) : "memory");
}
static __device__ __forceinline__ void wait_asynccnt_8() {
    asm volatile("s_wait_asynccnt 0x8" ::: "memory");
}
static __device__ __forceinline__ void wait_asynccnt_0() {
    asm volatile("s_wait_asynccnt 0x0" ::: "memory");
}

// ---------------------------------------------------------------------------
// Legacy A staging for the generic GEMM (dense packed layout, bf16):
//   sA[(((mt*KC + kc)*32 + L)*16 + h]  -> each lane's frag = contiguous 32B
//   A frag (16x32 MxK): lane L row M=L%16; half h -> K=(h/8)*16+(L/16)*8+h%8
// ---------------------------------------------------------------------------
static __device__ __forceinline__ void stage_A_packed(
    const float* __restrict__ A, int ldA, int K, int mBase,
    unsigned short* sA, int tid, int nthreads)
{
    const int KC = K >> 5;
    const int quads = (64 * K) >> 2;
    for (int q = tid; q < quads; q += nthreads) {
        int e0 = q << 2;
        int i  = e0 / K;
        int k0 = e0 - i * K;
        const float4 v = *(const float4*)(A + (size_t)(mBase + i) * ldA + k0);
        int mt = i >> 4, Lm = i & 15;
        int kc = k0 >> 5, km = k0 & 31;
        int hi = km >> 4, rest = km & 15;
        int L  = ((rest >> 3) << 4) | Lm;
        int h0 = (hi << 3) | (rest & 7);
        unsigned lo = f32_to_bf16(v.x) | (f32_to_bf16(v.y) << 16);
        unsigned hi2 = f32_to_bf16(v.z) | (f32_to_bf16(v.w) << 16);
        *(uint2*)((unsigned char*)sA + ((((mt * KC) + kc) << 5) + L) * 32 + h0 * 2)
            = make_uint2(lo, hi2);
    }
}

// ---------------------------------------------------------------------------
// Pack W[Kreal][Nreal] f32 -> bf16 B-fragments, zero-padded to (Kpad,Npad).
//   B frag (32x16 KxN): lane L col N=L%16; K=(h/8)*16+(L/16)*8+h%8
//   Chunk (kc,nt) = 1KB at ((kc*NT+nt)<<10), piece-split inside:
//     byte = chunk + (h>>3)*512 + L*16 + (h&7)*2
// ---------------------------------------------------------------------------
__global__ void pack_w_kernel(const float* __restrict__ W,
                              unsigned short* __restrict__ out,
                              int Kreal, int Nreal, int Kpad, int Npad)
{
    int idx = blockIdx.x * blockDim.x + threadIdx.x;
    int total = Kpad * Npad;
    if (idx >= total) return;
    int K = idx / Npad, N = idx - K * Npad;
    float v = (K < Kreal && N < Nreal) ? W[(size_t)K * Nreal + N] : 0.0f;
    int kc = K >> 5, km = K & 31;
    int nt = N >> 4, Ln = N & 15;
    int L  = (((km >> 3) & 1) << 4) | Ln;
    int h  = ((km >> 4) << 3) | (km & 7);
    int NT = Npad >> 4;
    size_t byte = ((size_t)(kc * NT + nt) << 10) + ((h >> 3) << 9) + (L << 4) + ((h & 7) << 1);
    out[byte >> 1] = (unsigned short)f32_to_bf16(v);
}

// load one B fragment given base byte pointer of its chunk piece-0
static __device__ __forceinline__ v16bf load_b_frag(const unsigned char* chunk, int lane) {
    union { v16bf v; v8bf h[2]; } bu;
    const unsigned char* c = chunk + lane * 16;
    bu.h[0] = *(const v8bf*)(c);
    bu.h[1] = *(const v8bf*)(c + B_PSTR);
    return bu.v;
}

// ---------------------------------------------------------------------------
// Generic WMMA GEMM: C = A(f32 MxK) @ Wpacked(bf16) + bias (f32 out, ldC, colOff)
// Block 128 thr (4 waves): wave w owns m-tile w; 8 n-tiles per blockIdx.y.
// ---------------------------------------------------------------------------
__global__ void __launch_bounds__(128)
gemm_wmma_kernel(const float* __restrict__ A, int ldA,
                 const unsigned short* __restrict__ Wp,
                 const float* __restrict__ bias,
                 float* __restrict__ C, int ldC, int colOff,
                 int K, int NT, int Nreal)
{
    extern __shared__ unsigned short sA[];
    const int tid = threadIdx.x;
    const int KC = K >> 5;
    const int mBase = blockIdx.x * 64;

    stage_A_packed(A, ldA, K, mBase, sA, tid, blockDim.x);
    __syncthreads();

    const int wave = tid >> 5, lane = tid & 31;
    const int mt = wave;
    const int ntBase = blockIdx.y * 8;
    const int ntEnd  = (ntBase + 8 < NT) ? (ntBase + 8) : NT;

    for (int nt = ntBase; nt < ntEnd; ++nt) {
        v8f acc = {};
        for (int kc = 0; kc < KC; ++kc) {
            v16bf a = *(const v16bf*)((const unsigned char*)sA
                        + ((((mt * KC) + kc) << 5) + lane) * 32);
            v16bf b = load_b_frag((const unsigned char*)Wp
                        + ((size_t)(kc * NT + nt) << 10), lane);
            acc = __builtin_amdgcn_wmma_f32_16x16x32_bf16(
                      false, a, false, b, (short)0, acc, false, false);
        }
        int col = (nt << 4) + (lane & 15);
        if (col < Nreal) {
            float bv = bias[col];
            int rowBase = mBase + (mt << 4) + ((lane >> 4) << 3);
#pragma unroll
            for (int r = 0; r < 8; ++r)
                C[(size_t)(rowBase + r) * ldC + colOff + col] = acc[r] + bv;
        }
    }
}

// ---------------------------------------------------------------------------
// Fused attention-score kernel:
//   scores[b,t] = sum_n tanh((enc[t,b,:]@W_ep)[n] + b_ep[n] + ha[b,n])*w_v[n] + b_v
// 64 enc rows (r = t*256+b) per block; K=N=512. A frags live in registers;
// B tiles are double-buffered in LDS via async global->LDS copies shared by
// all 4 waves. enc_attn (315 MB) is never materialized.
// ---------------------------------------------------------------------------
__global__ void __launch_bounds__(128)
attn_score_wmma_kernel(const float* __restrict__ enc,
                       const unsigned short* __restrict__ pWep,
                       const float* __restrict__ ha,
                       const float* __restrict__ b_ep,
                       const float* __restrict__ w_v,
                       const float* __restrict__ b_v,
                       float* __restrict__ scores_t /* [b*600 + t] */)
{
    extern __shared__ unsigned char smem[];
    unsigned char* sA = smem;              // A: 4 mt x 16 kc chunks, A_CSTR each
    unsigned char* sB = smem + A_REGION;   // B: 2 buffers x 16 kc chunks
    const int tid = threadIdx.x;
    const int NT = 32;
    const int mBase = blockIdx.x * 64;

    // ---- stage A (f32 -> bf16 swizzled, b64 stores) ----
    for (int q = tid; q < (64 * HIDDEN) >> 2; q += 128) {
        int e0 = q << 2;
        int i  = e0 >> 9;          // /512
        int k0 = e0 & 511;
        const float4 v = *(const float4*)(enc + (size_t)(mBase + i) * HIDDEN + k0);
        int mt = i >> 4, Lm = i & 15;
        int kc = k0 >> 5, km = k0 & 31;
        int hi = km >> 4, rest = km & 15;
        int L  = ((rest >> 3) << 4) | Lm;
        int h0 = (hi << 3) | (rest & 7);
        unsigned lo  = f32_to_bf16(v.x) | (f32_to_bf16(v.y) << 16);
        unsigned hi2 = f32_to_bf16(v.z) | (f32_to_bf16(v.w) << 16);
        *(uint2*)(sA + (mt * 16 + kc) * A_CSTR + L * 32 + h0 * 2) = make_uint2(lo, hi2);
    }

    const int wave = tid >> 5, lane = tid & 31;
    const int mt = wave;
    const unsigned sBlds0 = (unsigned)(unsigned long long)(sB);

    // ---- prologue: async-stage B tiles for nt=0 (each wave copies 4 kc chunks) ----
#pragma unroll
    for (int j = 0; j < 4; ++j) {
        int kc = wave * 4 + j;
        const unsigned char* g = (const unsigned char*)pWep
                                 + ((size_t)(kc * NT + 0) << 10) + lane * 16;
        unsigned l0 = sBlds0 + kc * B_CSTR + lane * 16;
        async_g2l_b128(l0,          g);
        async_g2l_b128(l0 + B_PSTR, g + 512);
    }

    __syncthreads();   // A staged

    // ---- hoist this wave's A fragments into registers (16 x 8 VGPRs) ----
    v16bf af[16];
#pragma unroll
    for (int kc = 0; kc < 16; ++kc)
        af[kc] = *(const v16bf*)(sA + (mt * 16 + kc) * A_CSTR + lane * 32);

    const int nsel = lane & 15;
    const int rowBase = mBase + (mt << 4) + ((lane >> 4) << 3);
    float sacc[8] = {0.f, 0.f, 0.f, 0.f, 0.f, 0.f, 0.f, 0.f};

    for (int nt = 0; nt < NT; ++nt) {
        int cur = nt & 1;
        if (nt + 1 < NT) {
            // issue next B tile into the other buffer, then wait for current
#pragma unroll
            for (int j = 0; j < 4; ++j) {
                int kc = wave * 4 + j;
                const unsigned char* g = (const unsigned char*)pWep
                                         + ((size_t)(kc * NT + nt + 1) << 10) + lane * 16;
                unsigned l0 = sBlds0 + (cur ^ 1) * B_BUF + kc * B_CSTR + lane * 16;
                async_g2l_b128(l0,          g);
                async_g2l_b128(l0 + B_PSTR, g + 512);
            }
            wait_asynccnt_8();
        } else {
            wait_asynccnt_0();
        }
        __syncthreads();   // current B buffer visible to all waves

        const unsigned char* bbuf = sB + cur * B_BUF;
        v8f acc = {};
#pragma unroll
        for (int kc = 0; kc < 16; ++kc) {
            v16bf b = load_b_frag(bbuf + kc * B_CSTR, lane);
            acc = __builtin_amdgcn_wmma_f32_16x16x32_bf16(
                      false, af[kc], false, b, (short)0, acc, false, false);
        }

        int n = (nt << 4) + nsel;
        float wv = w_v[n];
        float be = b_ep[n];
#pragma unroll
        for (int r = 0; r < 8; ++r) {
            int row = rowBase + r;
            int bb  = row & (BATCH - 1);
            sacc[r] += tanhf(acc[r] + be + ha[(size_t)bb * HIDDEN + n]) * wv;
        }
        __syncthreads();   // all waves done with this buffer before it is reused
    }

    // reduce over the 16 lanes of each half (xor bits 0..3 stay within a half)
#pragma unroll
    for (int r = 0; r < 8; ++r) {
        float v = sacc[r];
        v += __shfl_xor(v, 1, 32);
        v += __shfl_xor(v, 2, 32);
        v += __shfl_xor(v, 4, 32);
        v += __shfl_xor(v, 8, 32);
        sacc[r] = v;
    }
    float bv = b_v[0];
    if ((lane & 15) == 0) {
#pragma unroll
        for (int r = 0; r < 8; ++r) {
            int row = rowBase + r;
            int bb = row & (BATCH - 1);
            int t  = row >> 8;
            scores_t[(size_t)bb * TSEQ + t] = sacc[r] + bv;
        }
    }
}

// ---------------------------------------------------------------------------
// Softmax over T (600) per batch row; writes attn_weights (B,T).
// ---------------------------------------------------------------------------
__global__ void softmax_T_kernel(const float* __restrict__ s,
                                 float* __restrict__ attn)
{
    __shared__ float redmax[8];
    __shared__ float redsum[8];
    int b = blockIdx.x, tid = threadIdx.x;
    const float* row = s + (size_t)b * TSEQ;

    float mx = -INFINITY;
    for (int t = tid; t < TSEQ; t += 256) mx = fmaxf(mx, row[t]);
    for (int m = 16; m >= 1; m >>= 1) mx = fmaxf(mx, __shfl_xor(mx, m, 32));
    if ((tid & 31) == 0) redmax[tid >> 5] = mx;
    __syncthreads();
    mx = redmax[0];
#pragma unroll
    for (int i = 1; i < 8; ++i) mx = fmaxf(mx, redmax[i]);

    float sum = 0.f;
    for (int t = tid; t < TSEQ; t += 256) sum += __expf(row[t] - mx);
    for (int m = 16; m >= 1; m >>= 1) sum += __shfl_xor(sum, m, 32);
    if ((tid & 31) == 0) redsum[tid >> 5] = sum;
    __syncthreads();
    sum = 0.f;
#pragma unroll
    for (int i = 0; i < 8; ++i) sum += redsum[i];
    float inv = 1.f / sum;

    float* out = attn + (size_t)b * TSEQ;
    for (int t = tid; t < TSEQ; t += 256) out[t] = __expf(row[t] - mx) * inv;
}

// ---------------------------------------------------------------------------
// context[b,h] = sum_t attn[b,t] * enc[t,b,h]  — 315 MB streaming pass.
// ---------------------------------------------------------------------------
__global__ void context_kernel(const float* __restrict__ enc,
                               const float* __restrict__ attn,
                               float* __restrict__ ctx)
{
    __shared__ float w[TSEQ];
    int b = blockIdx.x;
    int h = blockIdx.y * 128 + threadIdx.x;
    for (int t = threadIdx.x; t < TSEQ; t += 128) w[t] = attn[(size_t)b * TSEQ + t];
    __syncthreads();

    const float* p = enc + (size_t)b * HIDDEN + h;
    const size_t stride = (size_t)BATCH * HIDDEN;
    float acc = 0.f;
    for (int t = 0; t < TSEQ; ++t) {
        if (t + 8 < TSEQ) __builtin_prefetch(p + (size_t)(t + 8) * stride, 0, 1);
        acc += w[t] * p[(size_t)t * stride];
    }
    ctx[(size_t)b * HIDDEN + h] = acc;
}

// ---------------------------------------------------------------------------
// argmax over vocab + embedding gather into in_dec[:,0:50]; zero cols 300..319.
// ---------------------------------------------------------------------------
__global__ void argmax_embed_kernel(const float* __restrict__ in_char,
                                    const float* __restrict__ emb,
                                    float* __restrict__ in_dec)
{
    int b = blockIdx.x, lane = threadIdx.x;
    float best = -INFINITY; int bi = 1 << 30;
    for (int v = lane; v < VOCAB; v += 32) {
        float x = in_char[(size_t)b * VOCAB + v];
        if (x > best) { best = x; bi = v; }
    }
    for (int m = 16; m >= 1; m >>= 1) {
        float ob = __shfl_xor(best, m, 32);
        int   oi = __shfl_xor(bi, m, 32);
        if (ob > best || (ob == best && oi < bi)) { best = ob; bi = oi; }
    }
    int top = __shfl(bi, 0, 32);
    for (int d = lane; d < EMBED; d += 32)
        in_dec[(size_t)b * 320 + d] = emb[(size_t)top * EMBED + d];
    for (int d = 300 + lane; d < 320; d += 32)
        in_dec[(size_t)b * 320 + d] = 0.f;
}

// ---------------------------------------------------------------------------
// GRU cell elementwise.
// ---------------------------------------------------------------------------
__global__ void gru_kernel(const float* __restrict__ gi,
                           const float* __restrict__ gh,
                           const float* __restrict__ h,
                           float* __restrict__ newh)
{
    int idx = blockIdx.x * 256 + threadIdx.x;
    if (idx >= BATCH * HIDDEN) return;
    int b = idx >> 9, j = idx & (HIDDEN - 1);
    const float* gib = gi + (size_t)b * (3 * HIDDEN);
    const float* ghb = gh + (size_t)b * (3 * HIDDEN);
    float i_r = gib[j], i_z = gib[HIDDEN + j], i_n = gib[2 * HIDDEN + j];
    float h_r = ghb[j], h_z = ghb[HIDDEN + j], h_n = ghb[2 * HIDDEN + j];
    float r = 1.f / (1.f + __expf(-(i_r + h_r)));
    float z = 1.f / (1.f + __expf(-(i_z + h_z)));
    float n = tanhf(i_n + r * h_n);
    float ho = h[idx];
    newh[idx] = (1.f - z) * n + z * ho;
}

// ---------------------------------------------------------------------------
// Softmax over vocab (83, padded logits ld=96). One wave per batch row.
// ---------------------------------------------------------------------------
__global__ void softmax_vocab_kernel(const float* __restrict__ logits,
                                     float* __restrict__ out)
{
    int b = blockIdx.x, lane = threadIdx.x;
    const float* row = logits + (size_t)b * 96;
    float mx = -INFINITY;
    for (int j = lane; j < VOCAB; j += 32) mx = fmaxf(mx, row[j]);
    for (int m = 16; m >= 1; m >>= 1) mx = fmaxf(mx, __shfl_xor(mx, m, 32));
    float sum = 0.f;
    for (int j = lane; j < VOCAB; j += 32) sum += __expf(row[j] - mx);
    for (int m = 16; m >= 1; m >>= 1) sum += __shfl_xor(sum, m, 32);
    float inv = 1.f / sum;
    for (int j = lane; j < VOCAB; j += 32)
        out[(size_t)b * VOCAB + j] = __expf(row[j] - mx) * inv;
}

// ---------------------------------------------------------------------------
// Workspace layout (bytes)
// ---------------------------------------------------------------------------
#define OFF_PWEP   0u
#define OFF_PWHP   524288u
#define OFF_PWHH   1048576u
#define OFF_PWIH   2621440u
#define OFF_PWCS   3604480u
#define OFF_PWOUT  3866624u
#define OFF_HA     3964928u
#define OFF_SC     4489216u
#define OFF_CTX    5103616u
#define OFF_INDEC  5627904u
#define OFF_GI     5955584u
#define OFF_GH     7528448u
#define OFF_LOG    9101312u

extern "C" void kernel_launch(void* const* d_in, const int* in_sizes, int n_in,
                              void* d_out, int out_size, void* d_ws, size_t ws_size,
                              hipStream_t stream)
{
    (void)in_sizes; (void)n_in; (void)out_size; (void)ws_size;

    const float* in_char = (const float*)d_in[0];
    const float* hidden  = (const float*)d_in[1];
    const float* enc     = (const float*)d_in[2];
    const float* W_hp    = (const float*)d_in[3];
    const float* b_hp    = (const float*)d_in[4];
    const float* W_ep    = (const float*)d_in[5];
    const float* b_ep    = (const float*)d_in[6];
    const float* w_v     = (const float*)d_in[7];
    const float* b_v     = (const float*)d_in[8];
    const float* W_cs    = (const float*)d_in[9];
    const float* b_cs    = (const float*)d_in[10];
    const float* emb     = (const float*)d_in[11];
    const float* W_ih    = (const float*)d_in[12];
    const float* b_ih    = (const float*)d_in[13];
    const float* W_hh    = (const float*)d_in[14];
    const float* b_hh    = (const float*)d_in[15];
    const float* W_out   = (const float*)d_in[16];
    const float* b_out   = (const float*)d_in[17];

    char* ws = (char*)d_ws;
    unsigned short* pWep  = (unsigned short*)(ws + OFF_PWEP);
    unsigned short* pWhp  = (unsigned short*)(ws + OFF_PWHP);
    unsigned short* pWhh  = (unsigned short*)(ws + OFF_PWHH);
    unsigned short* pWih  = (unsigned short*)(ws + OFF_PWIH);
    unsigned short* pWcs  = (unsigned short*)(ws + OFF_PWCS);
    unsigned short* pWout = (unsigned short*)(ws + OFF_PWOUT);
    float* ha      = (float*)(ws + OFF_HA);
    float* scores  = (float*)(ws + OFF_SC);
    float* ctx     = (float*)(ws + OFF_CTX);
    float* in_dec  = (float*)(ws + OFF_INDEC);
    float* gi      = (float*)(ws + OFF_GI);
    float* gh      = (float*)(ws + OFF_GH);
    float* logits  = (float*)(ws + OFF_LOG);

    float* out_probs = (float*)d_out;
    float* out_newh  = (float*)d_out + BATCH * VOCAB;
    float* out_attn  = (float*)d_out + BATCH * VOCAB + BATCH * HIDDEN;

    // Allow >64KB dynamic LDS on the score kernel. Idempotent host-side call,
    // executed unconditionally every invocation (no static state).
    hipFuncSetAttribute((const void*)attn_score_wmma_kernel,
                        hipFuncAttributeMaxDynamicSharedMemorySize, SCORE_LDS);

    // --- pack all weights to bf16 WMMA B-fragment order ---
    pack_w_kernel<<<(512 * 512 + 255) / 256, 256, 0, stream>>>(W_ep, pWep, 512, 512, 512, 512);
    pack_w_kernel<<<(512 * 512 + 255) / 256, 256, 0, stream>>>(W_hp, pWhp, 512, 512, 512, 512);
    pack_w_kernel<<<(512 * 1536 + 255) / 256, 256, 0, stream>>>(W_hh, pWhh, 512, 1536, 512, 1536);
    pack_w_kernel<<<(320 * 1536 + 255) / 256, 256, 0, stream>>>(W_ih, pWih, 300, 1536, 320, 1536);
    pack_w_kernel<<<(512 * 256 + 255) / 256, 256, 0, stream>>>(W_cs, pWcs, 512, 250, 512, 256);
    pack_w_kernel<<<(512 * 96 + 255) / 256, 256, 0, stream>>>(W_out, pWout, 512, 83, 512, 96);

    // --- argmax + embedding gather (independent) ---
    argmax_embed_kernel<<<BATCH, 32, 0, stream>>>(in_char, emb, in_dec);

    // --- hidden_attn = hidden @ W_hp + b_hp ---
    gemm_wmma_kernel<<<dim3(4, 4), 128, 64 * 512 * 2, stream>>>(
        hidden, 512, pWhp, b_hp, ha, 512, 0, 512, 32, 512);

    // --- fused attention scores (the 80.5 GFLOP GEMM + tanh + dot(w_v)) ---
    attn_score_wmma_kernel<<<(TSEQ * BATCH) / 64, 128, SCORE_LDS, stream>>>(
        enc, pWep, ha, b_ep, w_v, b_v, scores);

    // --- softmax over T -> attn_weights (output) ---
    softmax_T_kernel<<<BATCH, 256, 0, stream>>>(scores, out_attn);

    // --- context = sum_t attn * enc ---
    context_kernel<<<dim3(BATCH, 4), 128, 0, stream>>>(enc, out_attn, ctx);

    // --- in_dec[:,50:300] = ctx @ W_cs + b_cs ---
    gemm_wmma_kernel<<<dim3(4, 2), 128, 64 * 512 * 2, stream>>>(
        ctx, 512, pWcs, b_cs, in_dec, 320, 50, 512, 16, 250);

    // --- gi = in_dec @ W_ih + b_ih (K padded 300->320) ---
    gemm_wmma_kernel<<<dim3(4, 12), 128, 64 * 320 * 2, stream>>>(
        in_dec, 320, pWih, b_ih, gi, 1536, 0, 320, 96, 1536);

    // --- gh = h @ W_hh + b_hh ---
    gemm_wmma_kernel<<<dim3(4, 12), 128, 64 * 512 * 2, stream>>>(
        hidden, 512, pWhh, b_hh, gh, 1536, 0, 512, 96, 1536);

    // --- GRU cell -> new_h ---
    gru_kernel<<<(BATCH * HIDDEN) / 256, 256, 0, stream>>>(gi, gh, hidden, out_newh);

    // --- logits = new_h @ W_out + b_out ---
    gemm_wmma_kernel<<<dim3(4, 1), 128, 64 * 512 * 2, stream>>>(
        out_newh, 512, pWout, b_out, logits, 96, 0, 512, 6, 83);

    // --- output softmax ---
    softmax_vocab_kernel<<<BATCH, 32, 0, stream>>>(logits, out_probs);
}
